// JointLearningLayer_21088289423834
// MI455X (gfx1250) — compile-verified, matrix operand order checked by reading, not run on hardware
//
#include <hip/hip_runtime.h>

#define N_ 768
#define NN (768*768)
#define H_ 8
#define F_ 16
#define HF 128
#define E_ 24576
#define EIN 64
#define EOUT 64
#define SLOPE 0.2f
#define EPS 1e-5f

typedef __attribute__((ext_vector_type(16))) _Float16 v16h;
typedef __attribute__((ext_vector_type(8)))  float    v8f;

union Frag16 { v16h v; _Float16 h[16]; unsigned u[8]; uint4 q[2]; };

__device__ __forceinline__ float leakyf(float x) { return x >= 0.f ? x : SLOPE * x; }
__device__ __forceinline__ float eluf(float x)   { return x > 0.f ? x : __expf(x) - 1.f; }

// ---------------- workspace layout (float offsets) ----------------
#define WS_SE    0                         // 8*NN escores (leaky applied)
#define WS_SM    (WS_SE + 8*NN)            // NN softmax(mask) rows
#define WS_NP    (WS_SM + NN)              // N*HF np_proj
#define WS_SK    (WS_NP + N_*HF)           // N*HF skip
#define WS_SS    (WS_SK + N_*HF)           // N*H
#define WS_ST    (WS_SS + N_*H_)           // N*H
#define WS_ONP   (WS_ST + N_*H_)           // N*HF outn pre-BN
#define WS_OEP   (WS_ONP + N_*HF)          // E*EOUT oe pre-BN
#define WS_NSTAT (WS_OEP + E_*EOUT)        // 256: node BN mean[128] || var[128]
#define WS_ESTAT (WS_NSTAT + 256)          // 128: edge BN mean[64] || var[64]
#define WS_F16B  (WS_ESTAT + 128)          // f16 weights start (ushort units below)
#define WH_WP2   0                         // 128*64 halves
#define WH_WF4   (128*64)                  // 64*64 halves

// ---------------- output layout (float offsets) ----------------
#define OUT_OUTN 0
#define OUT_IE   (OUT_OUTN + N_*HF)
#define OUT_MASK (OUT_IE + (size_t)NN*EOUT)
#define OUT_OE   (OUT_MASK + NN)
#define OUT_IDX  (OUT_OE + (size_t)E_*EOUT)

// ---------------- f32 -> f16 weight conversion ----------------
__global__ void cvt_f16_kernel(const float* __restrict__ src,
                               unsigned short* __restrict__ dst, int n) {
  int t = blockIdx.x * blockDim.x + threadIdx.x;
  if (t < n) {
    union { _Float16 h; unsigned short u; } cv;
    cv.h = (_Float16)src[t];
    dst[t] = cv.u;
  }
}

// ---------------- row softmax of mask ----------------
__global__ __launch_bounds__(256) void mask_softmax_kernel(
    const float* __restrict__ mask, float* __restrict__ sm) {
  __shared__ float red[256];
  int i = blockIdx.x, t = threadIdx.x;
  float mx = -3.4e38f;
  for (int j = t; j < N_; j += 256) mx = fmaxf(mx, mask[(size_t)i*N_ + j]);
  red[t] = mx; __syncthreads();
  for (int s = 128; s > 0; s >>= 1) { if (t < s) red[t] = fmaxf(red[t], red[t+s]); __syncthreads(); }
  float rmax = red[0]; __syncthreads();
  float sum = 0.f;
  for (int j = t; j < N_; j += 256) sum += __expf(mask[(size_t)i*N_ + j] - rmax);
  red[t] = sum; __syncthreads();
  for (int s = 128; s > 0; s >>= 1) { if (t < s) red[t] += red[t+s]; __syncthreads(); }
  float rsum = red[0]; __syncthreads();
  float inv = 1.f / rsum;
  for (int j = t; j < N_; j += 256)
    sm[(size_t)i*N_ + j] = __expf(mask[(size_t)i*N_ + j] - rmax) * inv;
}

// ---------------- idx int -> float ----------------
__global__ void idx_cvt_kernel(const int* __restrict__ idx, float* __restrict__ out, int n) {
  int t = blockIdx.x * blockDim.x + threadIdx.x;
  if (t < n) out[t] = (float)idx[t];
}

// ---------------- node projections: np_proj = x@Wp.T, skip = x@Wskip.T ----------------
__global__ __launch_bounds__(256) void node_proj_kernel(
    const float* __restrict__ x, const float* __restrict__ Wp,
    const float* __restrict__ Wskip, float* __restrict__ np_proj,
    float* __restrict__ skip) {
  int t = blockIdx.x * blockDim.x + threadIdx.x;  // N*HF threads
  int r = t >> 7, c = t & 127;
  const float* xr = x + (size_t)r * HF;
  const float* wp = Wp + (size_t)c * HF;
  const float* wk = Wskip + (size_t)c * HF;
  float a = 0.f, b = 0.f;
  #pragma unroll 8
  for (int k = 0; k < HF; ++k) { float xv = xr[k]; a += xv * wp[k]; b += xv * wk[k]; }
  np_proj[t] = a; skip[t] = b;
}

// ---------------- ss/st score reductions ----------------
__global__ void score_kernel(const float* __restrict__ np_proj,
                             const float* __restrict__ ssrc,
                             const float* __restrict__ stgt,
                             float* __restrict__ ss, float* __restrict__ st) {
  int t = blockIdx.x * blockDim.x + threadIdx.x;  // N*H threads
  if (t >= N_*H_) return;
  int r = t >> 3, h = t & 7;
  const float* p = np_proj + (size_t)r * HF + h * F_;
  float a = 0.f, b = 0.f;
  #pragma unroll
  for (int f = 0; f < F_; ++f) { a += p[f] * ssrc[h*F_+f]; b += p[f] * stgt[h*F_+f]; }
  ss[t] = a; st[t] = b;
}

// ---- helper: build one 16x32 f16 A/B K-run (8 halves) from f32 via float4 pair ----
__device__ __forceinline__ void cvt8(Frag16& f, int base, float4 a, float4 b) {
  f.h[base+0] = (_Float16)a.x; f.h[base+1] = (_Float16)a.y;
  f.h[base+2] = (_Float16)a.z; f.h[base+3] = (_Float16)a.w;
  f.h[base+4] = (_Float16)b.x; f.h[base+5] = (_Float16)b.y;
  f.h[base+6] = (_Float16)b.z; f.h[base+7] = (_Float16)b.w;
}

// ---------------- big fused edge GEMM (WMMA f16): escores + ie ----------------
__global__ __launch_bounds__(256) void edge_gemm_wmma_kernel(
    const float* __restrict__ edges,           // (N,N,EIN)
    const unsigned short* __restrict__ wp2h,   // (HF,EIN) f16
    const unsigned short* __restrict__ wf4h,   // (EOUT,EIN) f16
    const float* __restrict__ bp2,             // HF
    const float* __restrict__ score_edge,      // HF
    const float* __restrict__ bf4,             // EOUT
    const float* __restrict__ sm,              // (N,N)
    float* __restrict__ se_out,                // (H,N,N) escores, leaky applied
    float* __restrict__ ie_out)                // (N,N,EOUT)
{
  __shared__ float stage[8][512];              // per-wave 2-tile staging (16KB)
  const int lane = threadIdx.x & 31;
  const int wave = threadIdx.x >> 5;
  const long tile = (long)blockIdx.x * 8 + wave;
  const long Rbase = tile * 16;                // flattened (i,j) row base
  const int i = (int)(Rbase / N_);
  const int jbase = (int)(Rbase % N_);
  const int mrow = lane & 15;
  const int hi = lane >> 4;                    // half-wave select
  const int n = lane & 15;

  // A fragments (16 rows x 64 K) built once, reused for all 12 col tiles.
  // Per lane the halves form contiguous 8-element K runs -> float4 loads.
  Frag16 a0, a1;
  {
    const float* arow = edges + (size_t)(Rbase + mrow) * EIN + hi * 8;
    cvt8(a0, 0, *(const float4*)(arow +  0), *(const float4*)(arow +  4));
    cvt8(a0, 8, *(const float4*)(arow + 16), *(const float4*)(arow + 20));
    cvt8(a1, 0, *(const float4*)(arow + 32), *(const float4*)(arow + 36));
    cvt8(a1, 8, *(const float4*)(arow + 48), *(const float4*)(arow + 52));
  }

  // --- 8 head tiles against Wp2, processed in pairs.
  // ep = leaky(D + bp2); stage ep*score in LDS; rowsum -> escore = leaky(sum)
  #pragma unroll
  for (int p = 0; p < 4; ++p) {
    const int ct0 = 2*p, ct1 = 2*p + 1;
    Frag16 b00, b01, b10, b11;   // {ct}{Kchunk}
    {
      const unsigned short* w0 = wp2h + (size_t)(ct0*16 + n) * EIN + hi * 8;
      const unsigned short* w1 = wp2h + (size_t)(ct1*16 + n) * EIN + hi * 8;
      b00.q[0] = *(const uint4*)(w0);      b00.q[1] = *(const uint4*)(w0 + 16);
      b01.q[0] = *(const uint4*)(w0 + 32); b01.q[1] = *(const uint4*)(w0 + 48);
      b10.q[0] = *(const uint4*)(w1);      b10.q[1] = *(const uint4*)(w1 + 16);
      b11.q[0] = *(const uint4*)(w1 + 32); b11.q[1] = *(const uint4*)(w1 + 48);
    }
    v8f acc0 = {0.f,0.f,0.f,0.f,0.f,0.f,0.f,0.f};
    v8f acc1 = {0.f,0.f,0.f,0.f,0.f,0.f,0.f,0.f};
    acc0 = __builtin_amdgcn_wmma_f32_16x16x32_f16(false, a0.v, false, b00.v, (short)0, acc0, false, false);
    acc0 = __builtin_amdgcn_wmma_f32_16x16x32_f16(false, a1.v, false, b01.v, (short)0, acc0, false, false);
    acc1 = __builtin_amdgcn_wmma_f32_16x16x32_f16(false, a0.v, false, b10.v, (short)0, acc1, false, false);
    acc1 = __builtin_amdgcn_wmma_f32_16x16x32_f16(false, a1.v, false, b11.v, (short)0, acc1, false, false);

    const float sc0 = score_edge[ct0*16 + n], bb0 = bp2[ct0*16 + n];
    const float sc1 = score_edge[ct1*16 + n], bb1 = bp2[ct1*16 + n];
    #pragma unroll
    for (int v = 0; v < 8; ++v) {
      stage[wave][      (v + hi*8)*16 + n] = leakyf(acc0[v] + bb0) * sc0;
      stage[wave][256 + (v + hi*8)*16 + n] = leakyf(acc1[v] + bb1) * sc1;
    }
    // Each lane reduces one of the 32 staged rows (tile = hi, row = n).
    const float* srow = &stage[wave][hi*256 + n*16];
    float4 q0 = *(const float4*)(srow + 0);
    float4 q1 = *(const float4*)(srow + 4);
    float4 q2 = *(const float4*)(srow + 8);
    float4 q3 = *(const float4*)(srow + 12);
    float s = (q0.x+q0.y+q0.z+q0.w) + (q1.x+q1.y+q1.z+q1.w)
            + (q2.x+q2.y+q2.z+q2.w) + (q3.x+q3.y+q3.z+q3.w);
    se_out[(size_t)(ct0 + hi)*NN + Rbase + n] = leakyf(s);
  }

  // --- 4 col tiles against Wf4: ie = (D + bf4) * softmax(mask) row factor
  float smrow[8];
  #pragma unroll
  for (int v = 0; v < 8; ++v)
    smrow[v] = sm[(size_t)i*N_ + jbase + v + hi*8];

  #pragma unroll
  for (int ct = 0; ct < 4; ++ct) {
    Frag16 b0, b1;
    {
      const unsigned short* wr = wf4h + (size_t)(ct*16 + n) * EIN + hi * 8;
      b0.q[0] = *(const uint4*)(wr);      b0.q[1] = *(const uint4*)(wr + 16);
      b1.q[0] = *(const uint4*)(wr + 32); b1.q[1] = *(const uint4*)(wr + 48);
    }
    v8f acc = {0.f,0.f,0.f,0.f,0.f,0.f,0.f,0.f};
    acc = __builtin_amdgcn_wmma_f32_16x16x32_f16(false, a0.v, false, b0.v, (short)0, acc, false, false);
    acc = __builtin_amdgcn_wmma_f32_16x16x32_f16(false, a1.v, false, b1.v, (short)0, acc, false, false);
    const float bb = bf4[ct*16 + n];
    #pragma unroll
    for (int v = 0; v < 8; ++v)
      ie_out[(size_t)(Rbase + v + hi*8)*EOUT + ct*16 + n] = (acc[v] + bb) * smrow[v];
  }
}

// ---------------- flash attention: outn_pre = softmax(scores+escores+mask) @ np_proj + skip --
__global__ __launch_bounds__(256) void attn_kernel(
    const float* __restrict__ ss, const float* __restrict__ st,
    const float* __restrict__ se,     // (H,N,N) escores (leaky applied)
    const float* __restrict__ mask,   // (N,N)
    const float* __restrict__ np_proj,// (N,HF)
    const float* __restrict__ skip,   // (N,HF)
    float* __restrict__ outn_pre)     // (N,HF)
{
  const int lane = threadIdx.x & 31;
  const int wave = threadIdx.x >> 5;
  const int p = blockIdx.x * 8 + wave;   // (i,h) pair
  const int h = p & 7;
  const int i = p >> 3;
  const int f = lane & 15;
  const int jpar = lane >> 4;

  const float ssv = ss[i*H_ + h];
  const float* seb = se + (size_t)h*NN + (size_t)i*N_;
  const float* mrow = mask + (size_t)i*N_;

  float m_run = -3.4e38f, s_run = 0.f, acc = 0.f;
  for (int j = jpar; j < N_; j += 2) {
    float l = leakyf(ssv + st[j*H_ + h]) + seb[j] + mrow[j];
    float nm = fmaxf(m_run, l);
    float eo = __expf(m_run - nm);
    float en = __expf(l - nm);
    s_run = s_run * eo + en;
    acc   = acc   * eo + en * np_proj[(size_t)j*HF + h*F_ + f];
    m_run = nm;
  }
  float m_o = __shfl_xor(m_run, 16);
  float s_o = __shfl_xor(s_run, 16);
  float a_o = __shfl_xor(acc,   16);
  float nm = fmaxf(m_run, m_o);
  float stot = s_run * __expf(m_run - nm) + s_o * __expf(m_o - nm);
  float atot = acc   * __expf(m_run - nm) + a_o * __expf(m_o - nm);
  if (lane < 16)
    outn_pre[(size_t)i*HF + h*F_ + f] = atot / stot + skip[(size_t)i*HF + h*F_ + f];
}

// ---------------- node batchnorm stats (one block per column) ----------------
__global__ __launch_bounds__(256) void node_bn_stats_kernel(
    const float* __restrict__ outn_pre, float* __restrict__ nstat) {
  __shared__ float r1[256], r2[256];
  int c = blockIdx.x, t = threadIdx.x;
  float s = 0.f, sq = 0.f;
  for (int r = t; r < N_; r += 256) { float v = outn_pre[(size_t)r*HF + c]; s += v; sq += v*v; }
  r1[t] = s; r2[t] = sq; __syncthreads();
  for (int k = 128; k > 0; k >>= 1) { if (t < k) { r1[t] += r1[t+k]; r2[t] += r2[t+k]; } __syncthreads(); }
  if (t == 0) {
    float mean = r1[0] / N_;
    nstat[c] = mean;
    nstat[HF + c] = r2[0] / N_ - mean*mean;
  }
}

// ---------------- node batchnorm apply + bias + elu -> final outn ----------------
__global__ __launch_bounds__(256) void node_bn_apply_kernel(
    const float* __restrict__ outn_pre, const float* __restrict__ nstat,
    const float* __restrict__ g, const float* __restrict__ b,
    const float* __restrict__ nbias, float* __restrict__ outn) {
  int t = blockIdx.x * blockDim.x + threadIdx.x;  // N*HF threads
  int c = t & 127;
  float mean = nstat[c], var = nstat[HF + c];
  outn[t] = eluf(g[c] * (outn_pre[t] - mean) * rsqrtf(var + EPS) + b[c] + nbias[c]);
}

// ---------------- sparse edge MLP -> oe_pre ----------------
__global__ __launch_bounds__(64) void sparse_edge_kernel(
    const float* __restrict__ sef, const int* __restrict__ idx,
    const float* __restrict__ outn,   // final outn (post-ELU)
    const float* __restrict__ Wf1, const float* __restrict__ bf1,
    const float* __restrict__ Wf2, const float* __restrict__ Wf3,
    const float* __restrict__ Wes, float* __restrict__ oe_pre) {
  __shared__ float sef_s[EIN];
  __shared__ float d2_s[HF];
  __shared__ float mr_s[2*EOUT];
  const int e = blockIdx.x, t = threadIdx.x;
  sef_s[t] = sef[(size_t)e*EIN + t];
  const int si = idx[e*2], sj = idx[e*2+1];
  #pragma unroll
  for (int q = t; q < HF; q += EIN) {
    float dv = outn[(size_t)si*HF + q] - outn[(size_t)sj*HF + q];
    d2_s[q] = dv * dv;
  }
  __syncthreads();
  float m = bf1[t];
  #pragma unroll 8
  for (int k = 0; k < EIN; ++k) m += sef_s[k] * Wf1[(size_t)t*EIN + k];
  mr_s[t] = leakyf(m);
  float r = 0.f;
  #pragma unroll 8
  for (int k = 0; k < HF; ++k) r += d2_s[k] * Wf2[(size_t)t*HF + k];
  mr_s[EOUT + t] = leakyf(r);
  __syncthreads();
  float o = 0.f;
  #pragma unroll 8
  for (int k = 0; k < 2*EOUT; ++k) o += mr_s[k] * Wf3[(size_t)t*2*EOUT + k];
  #pragma unroll 8
  for (int k = 0; k < EIN; ++k) o += sef_s[k] * Wes[(size_t)t*EIN + k];
  oe_pre[(size_t)e*EOUT + t] = o;
}

// ---------------- edge batchnorm stats ----------------
__global__ __launch_bounds__(256) void edge_bn_stats_kernel(
    const float* __restrict__ oe_pre, float* __restrict__ estat) {
  __shared__ float r1[256], r2[256];
  int c = blockIdx.x, t = threadIdx.x;
  float s = 0.f, sq = 0.f;
  for (int e = t; e < E_; e += 256) { float v = oe_pre[(size_t)e*EOUT + c]; s += v; sq += v*v; }
  r1[t] = s; r2[t] = sq; __syncthreads();
  for (int k = 128; k > 0; k >>= 1) { if (t < k) { r1[t] += r1[t+k]; r2[t] += r2[t+k]; } __syncthreads(); }
  if (t == 0) {
    float mean = r1[0] / E_;
    estat[c] = mean;
    estat[64 + c] = r2[0] / E_ - mean*mean;
  }
}

// ---------------- edge BN apply + elu -> oe, and scatter ie[si,sj] = oe ----------------
__global__ __launch_bounds__(256) void edge_bn_apply_kernel(
    const float* __restrict__ oe_pre, const float* __restrict__ estat,
    const float* __restrict__ g, const float* __restrict__ b,
    const int* __restrict__ idx, float* __restrict__ oe,
    float* __restrict__ ie) {
  int t = blockIdx.x * blockDim.x + threadIdx.x;  // E*EOUT threads
  int e = t >> 6, c = t & 63;
  float mean = estat[c], var = estat[64 + c];
  float v = eluf(g[c] * (oe_pre[t] - mean) * rsqrtf(var + EPS) + b[c]);
  oe[t] = v;
  int si = idx[e*2], sj = idx[e*2+1];
  ie[((size_t)si*N_ + sj)*EOUT + c] = v;
}

extern "C" void kernel_launch(void* const* d_in, const int* in_sizes, int n_in,
                              void* d_out, int out_size, void* d_ws, size_t ws_size,
                              hipStream_t stream) {
  const float* x          = (const float*)d_in[0];
  const float* edges      = (const float*)d_in[1];
  const float* mask       = (const float*)d_in[2];
  const float* sef        = (const float*)d_in[3];
  const int*   sidx       = (const int*)d_in[4];
  const float* Wp         = (const float*)d_in[5];
  const float* Wp2        = (const float*)d_in[6];
  const float* bp2        = (const float*)d_in[7];
  const float* score_src  = (const float*)d_in[8];
  const float* score_tgt  = (const float*)d_in[9];
  const float* score_edge = (const float*)d_in[10];
  const float* Wskip      = (const float*)d_in[11];
  const float* node_bias  = (const float*)d_in[12];
  const float* bn_node_g  = (const float*)d_in[13];
  const float* bn_node_b  = (const float*)d_in[14];
  const float* Wf1        = (const float*)d_in[15];
  const float* bf1        = (const float*)d_in[16];
  const float* Wf2        = (const float*)d_in[17];
  const float* Wf3        = (const float*)d_in[18];
  const float* Wf4        = (const float*)d_in[19];
  const float* bf4        = (const float*)d_in[20];
  const float* Wes        = (const float*)d_in[21];
  const float* bn_edge_g  = (const float*)d_in[22];
  const float* bn_edge_b  = (const float*)d_in[23];

  float* ws  = (float*)d_ws;
  float* out = (float*)d_out;
  unsigned short* f16b = (unsigned short*)(ws + WS_F16B);
  unsigned short* wp2h = f16b + WH_WP2;
  unsigned short* wf4h = f16b + WH_WF4;

  // 1. f16 weight conversion
  cvt_f16_kernel<<<(HF*EIN + 255)/256, 256, 0, stream>>>(Wp2, wp2h, HF*EIN);
  cvt_f16_kernel<<<(EOUT*EIN + 255)/256, 256, 0, stream>>>(Wf4, wf4h, EOUT*EIN);

  // 2. softmax(mask) rows
  mask_softmax_kernel<<<N_, 256, 0, stream>>>(mask, ws + WS_SM);

  // 3. pass-through outputs: mask copy, idx -> float
  hipMemcpyAsync(out + OUT_MASK, mask, (size_t)NN*sizeof(float),
                 hipMemcpyDeviceToDevice, stream);
  idx_cvt_kernel<<<(E_*2 + 255)/256, 256, 0, stream>>>(sidx, out + OUT_IDX, E_*2);

  // 4. node projections + attention scores
  node_proj_kernel<<<(N_*HF)/256, 256, 0, stream>>>(x, Wp, Wskip, ws + WS_NP, ws + WS_SK);
  score_kernel<<<(N_*H_ + 255)/256, 256, 0, stream>>>(ws + WS_NP, score_src, score_tgt,
                                                      ws + WS_SS, ws + WS_ST);

  // 5. big fused edge GEMM (WMMA): escores + ie (pre-scatter)
  edge_gemm_wmma_kernel<<<NN/16/8, 256, 0, stream>>>(
      edges, wp2h, wf4h, bp2, score_edge, bf4, ws + WS_SM,
      ws + WS_SE, out + OUT_IE);

  // 6. attention softmax + aggregation + skip
  attn_kernel<<<(N_*H_)/8, 256, 0, stream>>>(ws + WS_SS, ws + WS_ST, ws + WS_SE,
                                             mask, ws + WS_NP, ws + WS_SK, ws + WS_ONP);

  // 7. node batchnorm + bias + elu -> final outn
  node_bn_stats_kernel<<<HF, 256, 0, stream>>>(ws + WS_ONP, ws + WS_NSTAT);
  node_bn_apply_kernel<<<(N_*HF)/256, 256, 0, stream>>>(
      ws + WS_ONP, ws + WS_NSTAT, bn_node_g, bn_node_b, node_bias, out + OUT_OUTN);

  // 8. sparse edge MLP
  sparse_edge_kernel<<<E_, 64, 0, stream>>>(sef, sidx, out + OUT_OUTN,
                                            Wf1, bf1, Wf2, Wf3, Wes, ws + WS_OEP);

  // 9. edge batchnorm + elu + scatter into ie
  edge_bn_stats_kernel<<<EOUT, 256, 0, stream>>>(ws + WS_OEP, ws + WS_ESTAT);
  edge_bn_apply_kernel<<<(E_*EOUT)/256, 256, 0, stream>>>(
      ws + WS_OEP, ws + WS_ESTAT, bn_edge_g, bn_edge_b, sidx,
      out + OUT_OE, out + OUT_IE);
}
